// TransformerEncoderLayer_41549513621567
// MI455X (gfx1250) — matrix operand order repaired and checked by reading, not compiled, not run on hardware
//
#include <hip/hip_runtime.h>
#include <cstdint>
#include <cstddef>

// ---------------- problem constants (match reference) ----------------
#define BATCH   4
#define SEQ     2048
#define DMODEL  1024
#define NH      16
#define HD      64
#define DFF     4096
#define MROWS   (BATCH * SEQ)          // 8192
#define BIAS_W  (2 * SEQ - 1)          // 4095

// ---------------- types ----------------
typedef __bf16 bf16;
typedef __attribute__((ext_vector_type(8)))  bf16  bf16x8;
typedef __attribute__((ext_vector_type(16))) bf16  bf16x16;
typedef __attribute__((ext_vector_type(8)))  float f32x8;
typedef __attribute__((ext_vector_type(4)))  unsigned u32x4;
typedef __attribute__((ext_vector_type(8)))  int i32x8;
typedef __attribute__((ext_vector_type(4)))  int i32x4;

union AFrag { bf16x16 v; bf16x8 h[2]; };
union FAcc  { f32x8 v; float f[8]; };

__device__ __forceinline__ bf16 f2bf(float x) {
    union { float f; unsigned u; } in; in.f = x;
    unsigned u = in.u;
    u += 0x7fffu + ((u >> 16) & 1u);            // round-to-nearest-even
    union { unsigned short s; bf16 b; } out;
    out.s = (unsigned short)(u >> 16);
    return out.b;
}

// ---------------- CDNA5 data movers ----------------
// Tensor Data Mover: 2D tile (tileElems0 x tileRows), global row stride strideElems
// (elements are bf16: data_size=1 -> 2 bytes), LDS padding padAmt dwords every padInt dwords.
// pad codes: interval 0=2,1=4,2=8,3=16,4=32... dwords; amount 0=1,1=2,...  dwords.
__device__ __forceinline__ void tdm_load_tile(void* ldsDst, const void* gsrc,
                                              unsigned tileElems0, unsigned tileRows,
                                              unsigned strideElems,
                                              unsigned padIntCode, unsigned padAmtCode) {
    unsigned lds = (unsigned)(unsigned long long)ldsDst;          // low 32 bits = LDS offset
    unsigned long long ga = (unsigned long long)gsrc;
    u32x4 g0;
    g0[0] = 1u;                                  // count=1 (valid user descriptor)
    g0[1] = lds;                                 // lds_addr
    g0[2] = (unsigned)ga;                        // global_addr[31:0]
    g0[3] = (unsigned)(ga >> 32) | 0x80000000u;  // global_addr[56:32] | type=2 (bits 127:126)
    i32x8 g1;
    g1[0] = (int)((1u << 16)                     // data_size = 1 (2 bytes)
                | (1u << 20)                     // pad_enable
                | (padIntCode << 22)             // pad_interval
                | (padAmtCode << 25));           // pad_amount
    g1[1] = (int)(tileElems0 << 16);             // tensor_dim0[15:0] (bits 63:48)
    g1[2] = (int)(tileRows  << 16);              // tensor_dim1[15:0] (bits 95:80)
    g1[3] = (int)(tileElems0 << 16);             // tile_dim0 (bits 127:112)
    g1[4] = (int)tileRows;                       // tile_dim1 (tile_dim2 = 0)
    g1[5] = (int)strideElems;                    // tensor_dim0_stride[31:0]
    g1[6] = 0;
    g1[7] = 0;
    i32x4 z4 = {0, 0, 0, 0};
#if defined(__clang_major__) && (__clang_major__ >= 23)
    i32x8 z8 = {0, 0, 0, 0, 0, 0, 0, 0};
    __builtin_amdgcn_tensor_load_to_lds(g0, g1, z4, z4, z8, 0);
#else
    __builtin_amdgcn_tensor_load_to_lds(g0, g1, z4, z4, 0);
#endif
}

// Async global->LDS copy of 16 bytes per lane (no VGPR round trip; ASYNCcnt tracked).
__device__ __forceinline__ void async_ld_b128(void* ldsDst, const void* gsrc) {
    unsigned lds = (unsigned)(unsigned long long)ldsDst;
    unsigned long long ga = (unsigned long long)gsrc;
    asm volatile("global_load_async_to_lds_b128 %0, %1, off"
                 :: "v"(lds), "v"(ga) : "memory");
}

__device__ __forceinline__ void wait_async0() {
#if __has_builtin(__builtin_amdgcn_s_wait_asynccnt)
    __builtin_amdgcn_s_wait_asynccnt(0);
#else
    asm volatile("s_wait_asynccnt 0x0" ::: "memory");
#endif
}

// ---------------- fp32 -> bf16 convert ----------------
__global__ void __launch_bounds__(256)
f32_to_bf16_kernel(const float* __restrict__ in, bf16* __restrict__ out, size_t n) {
    size_t i = (size_t)blockIdx.x * 256 + threadIdx.x;
    if (i < n) out[i] = f2bf(in[i]);
}

// ---------------- T5 relative-position bias table: bias[h][(k-q)+S-1] ----------------
__global__ void __launch_bounds__(256)
build_bias_table(const float* __restrict__ rel_emb, float* __restrict__ biasT) {
    int idx = blockIdx.x * 256 + threadIdx.x;
    int total = BIAS_W * NH;
    if (idx >= total) return;
    int d = idx / NH - (SEQ - 1);                // relative position k - q
    int h = idx % NH;
    int n = -d;
    int ret = (n < 0) ? 16 : 0;
    n = (n < 0) ? -n : n;
    int bucket;
    if (n < 8) {
        bucket = ret + n;
    } else {
        int v = 8 + (int)(logf((float)n * (1.0f / 8.0f)) * (1.0f / logf(16.0f)) * 8.0f);
        bucket = ret + (v < 15 ? v : 15);
    }
    biasT[(size_t)h * BIAS_W + (d + SEQ - 1)] = rel_emb[bucket * NH + h];
}

// ---------------- layernorm (row of 1024) -> bf16 ----------------
__global__ void __launch_bounds__(256)
layernorm_bf16_kernel(const float* __restrict__ x, const float* __restrict__ g,
                      const float* __restrict__ beta, bf16* __restrict__ out) {
    __shared__ float redS[8], redQ[8];
    const int row = blockIdx.x;
    const float* xr = x + (size_t)row * DMODEL;
    float vals[4], s = 0.f, s2 = 0.f;
#pragma unroll
    for (int i = 0; i < 4; ++i) {
        float v = xr[threadIdx.x + i * 256];
        vals[i] = v; s += v; s2 += v * v;
    }
#pragma unroll
    for (int off = 1; off < 32; off <<= 1) {
        s  += __shfl_xor(s,  off, 32);
        s2 += __shfl_xor(s2, off, 32);
    }
    int w = threadIdx.x >> 5, lane = threadIdx.x & 31;
    if (lane == 0) { redS[w] = s; redQ[w] = s2; }
    __syncthreads();
    float ts = 0.f, tq = 0.f;
#pragma unroll
    for (int i = 0; i < 8; ++i) { ts += redS[i]; tq += redQ[i]; }
    float mu  = ts * (1.0f / DMODEL);
    float var = tq * (1.0f / DMODEL) - mu * mu;
    float inv = rsqrtf(var + 1e-6f);
#pragma unroll
    for (int i = 0; i < 4; ++i) {
        int c = threadIdx.x + i * 256;
        out[(size_t)row * DMODEL + c] = f2bf((vals[i] - mu) * inv * g[c] + beta[c]);
    }
}

// ---------------- WMMA GEMM: C[M,N] = A[M,K](bf16) x B[K,N](bf16) ----------------
// Double-buffered; A tile staged by the Tensor Data Mover (wave 0), B tile loaded to
// VGPRs before the WMMAs and transposed into LDS after them.
// mode 0: outF[m,n] = (acc+bias)*scale + residual
// mode 1: Q/K/V head scatter: outB[((b*NH+h)*SEQ+s)*HD+d] = bf16((acc+bias)*scale)
// mode 2: outB[m,n] = bf16(relu((acc+bias)*scale))
#define GTM 128
#define GTN 128
#define GTK 32
#define LDSTR 40                                  // 32 + 8 bf16 pad (80B rows, 16B aligned)

__global__ void __launch_bounds__(256)
gemm_bf16_wmma(const bf16* __restrict__ A, const bf16* __restrict__ B,
               const float* __restrict__ bias, const float* __restrict__ residual,
               float* __restrict__ outF, bf16* __restrict__ outB,
               int M, int N, int K, int mode, float scale) {
    __shared__ bf16 As[2][GTM * LDSTR];
    __shared__ bf16 Bs[2][GTN * LDSTR];           // transposed: [n][k]

    const int tid  = threadIdx.x;
    const int lane = tid & 31;
    const int w    = tid >> 5;                    // 0..7
    const int l    = lane & 15;
    const int half = lane >> 4;
    const int wm0  = (w >> 2) * 64;               // 2 waves in M
    const int wn0  = (w & 3) * 32;                // 4 waves in N
    const int m0   = blockIdx.y * GTM;
    const int n0   = blockIdx.x * GTN;
    const int KT   = K / GTK;

    // B staging coordinates for this thread (two bf16x8 chunks per k-step)
    const int kr0 = tid >> 4;                     // 0..15
    const int nc0 = (tid & 15) * 8;               // 0..120
    const int kr1 = kr0 + 16;                     // 16..31

    FAcc acc[4][2];
#pragma unroll
    for (int mt = 0; mt < 4; ++mt)
#pragma unroll
        for (int nt = 0; nt < 2; ++nt)
#pragma unroll
            for (int r = 0; r < 8; ++r) acc[mt][nt].f[r] = 0.f;

    // -------- prologue: stage tile 0 into buffer 0 --------
    if (w == 0)   // A tile 128x32, row stride K elems; pad 4 dwords per 16 dwords -> LDSTR
        tdm_load_tile(&As[0][0], A + (size_t)m0 * K, GTK, GTM, (unsigned)K, 3u, 3u);
    {
        bf16x8 v0 = *(const bf16x8*)(B + (size_t)kr0 * N + n0 + nc0);
        bf16x8 v1 = *(const bf16x8*)(B + (size_t)kr1 * N + n0 + nc0);
#pragma unroll
        for (int j = 0; j < 8; ++j) Bs[0][(nc0 + j) * LDSTR + kr0] = v0[j];
#pragma unroll
        for (int j = 0; j < 8; ++j) Bs[0][(nc0 + j) * LDSTR + kr1] = v1[j];
    }

    for (int kt = 0; kt < KT; ++kt) {
        const int cur = kt & 1;
        __builtin_amdgcn_s_wait_tensorcnt(0);     // my TDM (if any) has landed in LDS
        __syncthreads();                          // tile kt visible to all waves

        const bool hn = (kt + 1) < KT;
        bf16x8 nb0, nb1;
        if (hn) {
            const int k0n = (kt + 1) * GTK;
            if (w == 0)
                tdm_load_tile(&As[cur ^ 1][0], A + (size_t)m0 * K + k0n, GTK, GTM,
                              (unsigned)K, 3u, 3u);
            nb0 = *(const bf16x8*)(B + (size_t)(k0n + kr0) * N + n0 + nc0);
            nb1 = *(const bf16x8*)(B + (size_t)(k0n + kr1) * N + n0 + nc0);
        }

        AFrag a[4], bf[2];
#pragma unroll
        for (int mt = 0; mt < 4; ++mt) {
            int row = wm0 + mt * 16 + l;
            a[mt].h[0] = *(const bf16x8*)(&As[cur][row * LDSTR + half * 8]);
            a[mt].h[1] = *(const bf16x8*)(&As[cur][row * LDSTR + 16 + half * 8]);
        }
#pragma unroll
        for (int nt = 0; nt < 2; ++nt) {
            int col = wn0 + nt * 16 + l;
            bf[nt].h[0] = *(const bf16x8*)(&Bs[cur][col * LDSTR + half * 8]);
            bf[nt].h[1] = *(const bf16x8*)(&Bs[cur][col * LDSTR + 16 + half * 8]);
        }
#pragma unroll
        for (int mt = 0; mt < 4; ++mt)
#pragma unroll
            for (int nt = 0; nt < 2; ++nt)
                acc[mt][nt].v = __builtin_amdgcn_wmma_f32_16x16x32_bf16(
                    false, a[mt].v, false, bf[nt].v, (short)0, acc[mt][nt].v, false, false);

        if (hn) {   // transpose-store next B tile while WMMAs drain
#pragma unroll
            for (int j = 0; j < 8; ++j) Bs[cur ^ 1][(nc0 + j) * LDSTR + kr0] = nb0[j];
#pragma unroll
            for (int j = 0; j < 8; ++j) Bs[cur ^ 1][(nc0 + j) * LDSTR + kr1] = nb1[j];
        }
    }

    // -------- epilogue --------
#pragma unroll
    for (int mt = 0; mt < 4; ++mt) {
#pragma unroll
        for (int nt = 0; nt < 2; ++nt) {
            int n = n0 + wn0 + nt * 16 + l;
            float bia = bias ? bias[n] : 0.f;
#pragma unroll
            for (int r = 0; r < 8; ++r) {
                int m = m0 + wm0 + mt * 16 + r + half * 8;
                float v = (acc[mt][nt].f[r] + bia) * scale;
                if (mode == 0) {
                    if (residual) v += residual[(size_t)m * N + n];
                    outF[(size_t)m * N + n] = v;
                } else if (mode == 1) {
                    int bb = m / SEQ, ss = m % SEQ;
                    int hh = n >> 6, dd = n & 63;
                    outB[(((size_t)bb * NH + hh) * SEQ + ss) * HD + dd] = f2bf(v);
                } else {
                    v = v > 0.f ? v : 0.f;
                    outB[(size_t)m * N + n] = f2bf(v);
                }
            }
        }
    }
}

// ---------------- flash attention: block = 128 q rows (8 waves x 16), stream 32 keys/iter ----------------
__global__ void __launch_bounds__(256)
attention_fwd(const bf16* __restrict__ qh, const bf16* __restrict__ kh,
              const bf16* __restrict__ vh, const float* __restrict__ biasT,
              bf16* __restrict__ ctx) {
    __shared__ bf16 Ks[32 * 72];                  // [key][d], stride 72 (144B rows)
    __shared__ bf16 Vt[64 * 40];                  // [d][key], stride 40
    __shared__ bf16 Pw[8 * 16 * 40];              // per-wave P [16 q][32 k], stride 40

    const int tid  = threadIdx.x;
    const int lane = tid & 31;
    const int w    = tid >> 5;
    const int l    = lane & 15;
    const int half = lane >> 4;
    const int h    = blockIdx.y;
    const int b    = blockIdx.z;
    const int q0   = blockIdx.x * 128;

    const size_t bh = (size_t)b * NH + h;
    const bf16* qbase = qh + bh * SEQ * HD;
    const bf16* kbase = kh + bh * SEQ * HD;
    const bf16* vbase = vh + bh * SEQ * HD;
    const float* brow = biasT + (size_t)h * BIAS_W + (SEQ - 1);

    // Q A-fragments (2 d-chunks of 32); Q already scaled by 1/sqrt(HD)
    AFrag qa[2];
    {
        int row = q0 + w * 16 + l;
        const bf16* qp = qbase + (size_t)row * HD;
#pragma unroll
        for (int c = 0; c < 2; ++c) {
            qa[c].h[0] = *(const bf16x8*)(qp + c * 32 + half * 8);
            qa[c].h[1] = *(const bf16x8*)(qp + c * 32 + 16 + half * 8);
        }
    }

    FAcc cacc[4];
    float mrow[8], lrow[8];
#pragma unroll
    for (int dt = 0; dt < 4; ++dt)
#pragma unroll
        for (int r = 0; r < 8; ++r) cacc[dt].f[r] = 0.f;
#pragma unroll
    for (int r = 0; r < 8; ++r) { mrow[r] = -1e30f; lrow[r] = 0.f; }

    for (int kb = 0; kb < SEQ; kb += 32) {
        __syncthreads();
        {   // K tile: async copy straight to LDS; V tile: load + transpose-store
            int key = tid >> 3, dc = (tid & 7) * 8;
            async_ld_b128(&Ks[key * 72 + dc], kbase + (size_t)(kb + key) * HD + dc);
            bf16x8 vv = *(const bf16x8*)(vbase + (size_t)(kb + key) * HD + dc);
#pragma unroll
            for (int j = 0; j < 8; ++j) Vt[(dc + j) * 40 + key] = vv[j];
        }
        wait_async0();
        __syncthreads();

        // scores: 2 tiles of 16 keys, each = 2 chained WMMAs over d
        FAcc s[2];
#pragma unroll
        for (int t = 0; t < 2; ++t) {
            AFrag kf0, kf1;
            int key = t * 16 + l;
            kf0.h[0] = *(const bf16x8*)(&Ks[key * 72 + half * 8]);
            kf0.h[1] = *(const bf16x8*)(&Ks[key * 72 + 16 + half * 8]);
            kf1.h[0] = *(const bf16x8*)(&Ks[key * 72 + 32 + half * 8]);
            kf1.h[1] = *(const bf16x8*)(&Ks[key * 72 + 48 + half * 8]);
            FAcc z;
#pragma unroll
            for (int r = 0; r < 8; ++r) z.f[r] = 0.f;
            s[t].v = __builtin_amdgcn_wmma_f32_16x16x32_bf16(
                false, qa[0].v, false, kf0.v, (short)0, z.v, false, false);
            s[t].v = __builtin_amdgcn_wmma_f32_16x16x32_bf16(
                false, qa[1].v, false, kf1.v, (short)0, s[t].v, false, false);
#pragma unroll
            for (int r = 0; r < 8; ++r) {
                int qrow = q0 + w * 16 + r + half * 8;
                int krow = kb + t * 16 + l;
                s[t].f[r] += brow[krow - qrow];
            }
        }

        // online softmax over the 32 new keys
        float mnew[8], fsc[8];
#pragma unroll
        for (int r = 0; r < 8; ++r) {
            float mx = fmaxf(s[0].f[r], s[1].f[r]);
#pragma unroll
            for (int off = 1; off < 16; off <<= 1)
                mx = fmaxf(mx, __shfl_xor(mx, off, 32));
            mnew[r] = fmaxf(mrow[r], mx);
            fsc[r]  = __expf(mrow[r] - mnew[r]);
            mrow[r] = mnew[r];
        }
#pragma unroll
        for (int dt = 0; dt < 4; ++dt)
#pragma unroll
            for (int r = 0; r < 8; ++r) cacc[dt].f[r] *= fsc[r];
#pragma unroll
        for (int r = 0; r < 8; ++r) {
            float p0 = __expf(s[0].f[r] - mnew[r]);
            float p1 = __expf(s[1].f[r] - mnew[r]);
            float rs = p0 + p1;
#pragma unroll
            for (int off = 1; off < 16; off <<= 1)
                rs += __shfl_xor(rs, off, 32);
            lrow[r] = lrow[r] * fsc[r] + rs;
            int prow = r + half * 8;                       // D-tile row -> P row
            Pw[(w * 16 + prow) * 40 + l]      = f2bf(p0);
            Pw[(w * 16 + prow) * 40 + 16 + l] = f2bf(p1);
        }
        __syncthreads();

        // P (16x32) as A-fragment, V columns as B-fragments: ctx += P x V
        AFrag pa;
        {
            const bf16* pr = &Pw[(w * 16 + l) * 40];
            pa.h[0] = *(const bf16x8*)(pr + half * 8);
            pa.h[1] = *(const bf16x8*)(pr + 16 + half * 8);
        }
#pragma unroll
        for (int dt = 0; dt < 4; ++dt) {
            AFrag vf;
            int d = dt * 16 + l;
            vf.h[0] = *(const bf16x8*)(&Vt[d * 40 + half * 8]);
            vf.h[1] = *(const bf16x8*)(&Vt[d * 40 + 16 + half * 8]);
            cacc[dt].v = __builtin_amdgcn_wmma_f32_16x16x32_bf16(
                false, pa.v, false, vf.v, (short)0, cacc[dt].v, false, false);
        }
    }

    // normalize and write ctx as bf16 in [B*S, DMODEL] (cols h*64+d)
#pragma unroll
    for (int dt = 0; dt < 4; ++dt) {
#pragma unroll
        for (int r = 0; r < 8; ++r) {
            int qrow = q0 + w * 16 + r + half * 8;
            int col  = h * HD + dt * 16 + l;
            ctx[((size_t)b * SEQ + qrow) * DMODEL + col] = f2bf(cacc[dt].f[r] / lrow[r]);
        }
    }
}

// ---------------- host orchestration ----------------
extern "C" void kernel_launch(void* const* d_in, const int* in_sizes, int n_in,
                              void* d_out, int out_size, void* d_ws, size_t ws_size,
                              hipStream_t stream) {
    (void)in_sizes; (void)n_in; (void)out_size; (void)ws_size;
    const float* src   = (const float*)d_in[0];
    const float* wq    = (const float*)d_in[1];
    const float* bq    = (const float*)d_in[2];
    const float* wk    = (const float*)d_in[3];
    const float* bk    = (const float*)d_in[4];
    const float* wv    = (const float*)d_in[5];
    const float* bv    = (const float*)d_in[6];
    const float* wo    = (const float*)d_in[7];
    const float* bo    = (const float*)d_in[8];
    const float* w1    = (const float*)d_in[9];
    const float* b1    = (const float*)d_in[10];
    const float* w2    = (const float*)d_in[11];
    const float* b2    = (const float*)d_in[12];
    const float* ln1g  = (const float*)d_in[13];
    const float* ln1b  = (const float*)d_in[14];
    const float* ln2g  = (const float*)d_in[15];
    const float* ln2b  = (const float*)d_in[16];
    const float* rele  = (const float*)d_in[17];
    float* out = (float*)d_out;

    char* ws = (char*)d_ws;
    size_t off = 0;
    auto take = [&](size_t bytes) { void* p = ws + off; off += (bytes + 255) & ~(size_t)255; return p; };

    bf16* wqb  = (bf16*)take((size_t)DMODEL * DMODEL * 2);
    bf16* wkb  = (bf16*)take((size_t)DMODEL * DMODEL * 2);
    bf16* wvb  = (bf16*)take((size_t)DMODEL * DMODEL * 2);
    bf16* wob  = (bf16*)take((size_t)DMODEL * DMODEL * 2);
    bf16* w1b  = (bf16*)take((size_t)DMODEL * DFF * 2);
    bf16* w2b  = (bf16*)take((size_t)DFF * DMODEL * 2);
    bf16* xnb  = (bf16*)take((size_t)MROWS * DMODEL * 2);
    bf16* qhb  = (bf16*)take((size_t)MROWS * DMODEL * 2);
    bf16* khb  = (bf16*)take((size_t)MROWS * DMODEL * 2);
    bf16* vhb  = (bf16*)take((size_t)MROWS * DMODEL * 2);
    bf16* ctxb = (bf16*)take((size_t)MROWS * DMODEL * 2);
    float* xf  = (float*)take((size_t)MROWS * DMODEL * 4);
    bf16* ynb  = (bf16*)take((size_t)MROWS * DMODEL * 2);
    bf16* hb   = (bf16*)take((size_t)MROWS * DFF * 2);
    float* biasT = (float*)take((size_t)NH * BIAS_W * 4);

    const size_t nDD = (size_t)DMODEL * DMODEL;
    const size_t nDF = (size_t)DMODEL * DFF;
    f32_to_bf16_kernel<<<(nDD + 255) / 256, 256, 0, stream>>>(wq, wqb, nDD);
    f32_to_bf16_kernel<<<(nDD + 255) / 256, 256, 0, stream>>>(wk, wkb, nDD);
    f32_to_bf16_kernel<<<(nDD + 255) / 256, 256, 0, stream>>>(wv, wvb, nDD);
    f32_to_bf16_kernel<<<(nDD + 255) / 256, 256, 0, stream>>>(wo, wob, nDD);
    f32_to_bf16_kernel<<<(nDF + 255) / 256, 256, 0, stream>>>(w1, w1b, nDF);
    f32_to_bf16_kernel<<<(nDF + 255) / 256, 256, 0, stream>>>(w2, w2b, nDF);

    build_bias_table<<<(NH * BIAS_W + 255) / 256, 256, 0, stream>>>(rele, biasT);

    // LN1 -> xn (bf16)
    layernorm_bf16_kernel<<<MROWS, 256, 0, stream>>>(src, ln1g, ln1b, xnb);

    // Q/K/V projections into head layout; Q pre-scaled by 1/sqrt(HD)
    dim3 gQKV(DMODEL / GTN, MROWS / GTM);
    gemm_bf16_wmma<<<gQKV, 256, 0, stream>>>(xnb, wqb, bq, nullptr, nullptr, qhb,
                                             MROWS, DMODEL, DMODEL, 1, 0.125f);
    gemm_bf16_wmma<<<gQKV, 256, 0, stream>>>(xnb, wkb, bk, nullptr, nullptr, khb,
                                             MROWS, DMODEL, DMODEL, 1, 1.0f);
    gemm_bf16_wmma<<<gQKV, 256, 0, stream>>>(xnb, wvb, bv, nullptr, nullptr, vhb,
                                             MROWS, DMODEL, DMODEL, 1, 1.0f);

    // flash attention -> ctx (bf16, [B*S, DMODEL])
    dim3 gAtt(SEQ / 128, NH, BATCH);
    attention_fwd<<<gAtt, 256, 0, stream>>>(qhb, khb, vhb, biasT, ctxb);

    // out projection + residual(src) -> x (f32)
    gemm_bf16_wmma<<<gQKV, 256, 0, stream>>>(ctxb, wob, bo, src, xf, nullptr,
                                             MROWS, DMODEL, DMODEL, 0, 1.0f);

    // LN2 -> yn (bf16)
    layernorm_bf16_kernel<<<MROWS, 256, 0, stream>>>(xf, ln2g, ln2b, ynb);

    // FFN1 (+ReLU) -> h (bf16)
    dim3 gF1(DFF / GTN, MROWS / GTM);
    gemm_bf16_wmma<<<gF1, 256, 0, stream>>>(ynb, w1b, b1, nullptr, nullptr, hb,
                                            MROWS, DFF, DMODEL, 2, 1.0f);

    // FFN2 + residual(x) -> out (f32)
    dim3 gF2(DMODEL / GTN, MROWS / GTM);
    gemm_bf16_wmma<<<gF2, 256, 0, stream>>>(hb, w2b, b2, xf, out, nullptr,
                                            MROWS, DMODEL, DFF, 0, 1.0f);
}
